// Spinn_84189948936632
// MI455X (gfx1250) — compile-verified, matrix operand order checked by reading, not run on hardware
//
#include <hip/hip_runtime.h>
#include <hip/hip_bf16.h>

// ---------------------------------------------------------------------------
// SPINN TreeLSTM scan for MI455X (gfx1250).
//
// 512 strictly sequential steps; per step: gemm [256,320]x[320,640] + LSTM
// cell with stack-gathered children (~105 MFLOP/step). Latency/sync-bound =>
// persistent single-WGP kernel (16 waves, workgroup barrier per step),
// bf16 WMMA 16x16x32, A staged in LDS (~164KB of the 320KB WGP LDS),
// W packed once into per-lane WMMA B-fragment layout (hot in L2/WGP$).
//
// Round-2 fix: pre-convert the whole embedding table to bf16 once (setup
// kernel, full-chip parallel) so the per-step A-build is pure bf16 copies —
// no f32->bf16 conversion VALU work inside the 512-step sequential loop.
// ---------------------------------------------------------------------------

#define D_STEPS 512
#define NB      256          // batch
#define HD      128          // hidden
#define LD      64           // label dim
#define VOCAB   32000
#define KDIM    320          // L + 2H
#define GN      640          // 5H
#define NKT     10           // KDIM / 32
#define NNT     40           // GN / 16
#define ASTRIDE 328          // padded LDS row stride (bf16 elems)

typedef __attribute__((ext_vector_type(16))) __bf16 v16bf;
typedef __attribute__((ext_vector_type(8)))  __bf16 v8bf;
typedef __attribute__((ext_vector_type(8)))  float  v8f;

// workspace layout (bytes); total ~101 MB
#define WP_OFF   0u            // packed W: 10*40*32*16 bf16 = 409600 B
#define EMB_OFF  524288u       // bf16 emb: 32000*64*2 = 4096000 B
#define HG_OFF   4718592u      // h history: 512*256*128 bf16 = 33554432 B
#define CG_OFF   38273024u     // c history: 512*256*128 f32  = 67108864 B
#define IDX_OFF  105381888u    // thin-stack indices: 512*256 i32 = 524288 B

__device__ __forceinline__ float sigf(float x) {
  return 1.0f / (1.0f + __expf(-x));
}

// Pack W [320,640] f32 -> bf16 in per-lane WMMA B-fragment order.
// Fragment element e of lane l (khalf = l>>4):
//   k_local = (e<8) ? khalf*8 + e : 16 + khalf*8 + (e-8);  col = nt*16 + (l&15)
// mirroring the 16-bit A-matrix VGPR layout (CDNA5 ISA §7.12.2).
__global__ void spinn_pack_w(const float* __restrict__ W, __bf16* __restrict__ Wp) {
  int idx = blockIdx.x * blockDim.x + threadIdx.x;
  if (idx >= NKT * NNT * 32 * 16) return;
  int e  = idx & 15;
  int l  = (idx >> 4) & 31;
  int nt = (idx >> 9) % NNT;
  int kt = idx / (16 * 32 * NNT);
  int khalf = l >> 4;
  int kl = (e < 8) ? (khalf * 8 + e) : (16 + khalf * 8 + (e - 8));
  int k  = kt * 32 + kl;
  int col = nt * 16 + (l & 15);
  Wp[idx] = (__bf16)W[k * GN + col];
}

// One-time f32 -> bf16 conversion of the embedding table (vectorized x4).
__global__ void spinn_cvt_emb(const float* __restrict__ emb,
                              __bf16* __restrict__ embb) {
  int idx = blockIdx.x * blockDim.x + threadIdx.x;   // over float4 groups
  if (idx >= VOCAB * LD / 4) return;
  float4 v = ((const float4*)emb)[idx];
  __bf16* d = embb + (size_t)idx * 4;
  d[0] = (__bf16)v.x; d[1] = (__bf16)v.y;
  d[2] = (__bf16)v.z; d[3] = (__bf16)v.w;
}

__global__ __launch_bounds__(512, 1)
void spinn_persistent(const int* __restrict__ trans,
                      const int* __restrict__ labels,
                      const __bf16* __restrict__ embb,
                      const float* __restrict__ bias,
                      const __bf16* __restrict__ Wp,
                      __bf16* __restrict__ h_g,
                      float* __restrict__ c_g,
                      int* __restrict__ idxs,
                      float* __restrict__ out) {
  extern __shared__ __align__(16) char smem[];
  __bf16* A = (__bf16*)smem;                        // [256][ASTRIDE] bf16
  int* lsel = (int*)(smem + NB * ASTRIDE * 2);      // left-child step (-1 = leaf)
  int* rsel = lsel + NB;
  int* sptr = rsel + NB;                            // per-batch stack pointer

  const int tid   = threadIdx.x;
  const int wave  = tid >> 5;
  const int lid   = tid & 31;
  const int lrow  = lid & 15;     // A row within tile / D column
  const int khalf = lid >> 4;     // which K half of the fragment this lane holds

  // Per-wave invariants: group id g = wave + 16*j  =>  hb = g & 7 = wave & 7
  // is FIXED per wave; only mt = g >> 3 varies. Hoist everything hb-derived.
  const int hb   = wave & 7;
  const int colh = hb * 16 + lrow;                  // output h-column
  const float bi  = bias[colh];
  const float bfl = bias[128 + colh];
  const float bfr = bias[256 + colh];
  const float bo  = bias[384 + colh];
  const float bu  = bias[512 + colh];
  const __bf16* wbase = Wp + ((size_t)hb * 32 + lid) * 16;  // + kt*NNT*512 + gate*8*512

  if (tid < NB) sptr[tid] = 0;

  // Warm L2/WGP$ with the packed weights once; they stay hot for all 512 steps.
  for (int off = tid * 64; off < NKT * NNT * 32 * 16; off += 512 * 64)
    __builtin_prefetch(Wp + off, 0, 3);
  __syncthreads();

  float* outc = out;              // reference returns (c[D-1], h[D-1])
  float* outh = out + NB * HD;

  for (int t = 0; t < D_STEPS; ++t) {
    // ---- phase 1: thin-stack bookkeeping (one thread per batch column) ----
    if (tid < NB) {
      int n = tid;
      int m = trans[t * NB + n];          // 1 = reduce, 0 = shift
      int p = sptr[n];
      int ls = -1, rs = -1;
      if (m) {                            // schedule guarantees p >= 2 here
        rs = idxs[(p - 1) * NB + n];      // right popped first
        ls = idxs[(p - 2) * NB + n];
      }
      lsel[n] = ls;
      rsel[n] = rs;
      int np = p - 2 * m;
      idxs[np * NB + n] = t;              // push current step for all columns
      sptr[n] = np + 1;
    }
    __syncthreads();

    // ---- phase 2: build A = [emb(lab) | h_left | h_right] bf16 in LDS ----
    // Pure bf16 16B copies; no conversions on the sequential critical path.
    {
      int row = tid >> 1, q = tid & 1;
      int lab = labels[t * NB + row];
      const uint4* esrc = (const uint4*)(embb + (size_t)lab * LD + q * 32);
      uint4* edst = (uint4*)(A + row * ASTRIDE + q * 32);
      #pragma unroll
      for (int j = 0; j < 4; ++j) edst[j] = esrc[j];
      // child h gather (bf16 copy) or leaf zeros; 128 bf16 = 16 x uint4
      int sel = q ? rsel[row] : lsel[row];
      uint4* hdst = (uint4*)(A + row * ASTRIDE + LD + q * HD);
      if (sel >= 0) {
        const uint4* hsrc = (const uint4*)(h_g + ((size_t)sel * NB + row) * HD);
        #pragma unroll 4
        for (int j = 0; j < 16; ++j) hdst[j] = hsrc[j];
      } else {
        uint4 z = {0u, 0u, 0u, 0u};
        #pragma unroll 4
        for (int j = 0; j < 16; ++j) hdst[j] = z;
      }
    }
    __syncthreads();

    // ---- phase 3: WMMA gemm + fused TreeLSTM cell ----
    // Wave handles groups (mt, hb) with fixed hb; 5 gate tiles per group
    // share the A fragment per k-step.
    for (int g = wave; g < 128; g += 16) {
      int mt = g >> 3;
      v8f acc0 = {}, acc1 = {}, acc2 = {}, acc3 = {}, acc4 = {};
      const __bf16* arow = A + (mt * 16 + lrow) * ASTRIDE + khalf * 8;
      // unroll 2 (NOT full): keeps <=2 A frags + 10 B frags + 5 accums live
      // (~150 VGPRs) so nothing spills, while allowing load/WMMA overlap.
      #pragma unroll 2
      for (int kt = 0; kt < NKT; ++kt) {
        v8bf alo = *(const v8bf*)(arow + kt * 32);
        v8bf ahi = *(const v8bf*)(arow + kt * 32 + 16);
        v16bf af = __builtin_shufflevector(alo, ahi,
            0, 1, 2, 3, 4, 5, 6, 7, 8, 9, 10, 11, 12, 13, 14, 15);
        const __bf16* bb = wbase + (size_t)kt * NNT * 32 * 16;
        v16bf b0 = *(const v16bf*)(bb);                    // i  : nt = hb
        v16bf b1 = *(const v16bf*)(bb + 8 * 32 * 16);      // fl : nt = 8+hb
        v16bf b2 = *(const v16bf*)(bb + 16 * 32 * 16);     // fr
        v16bf b3 = *(const v16bf*)(bb + 24 * 32 * 16);     // o
        v16bf b4 = *(const v16bf*)(bb + 32 * 32 * 16);     // u
        acc0 = __builtin_amdgcn_wmma_f32_16x16x32_bf16(false, af, false, b0,
                 (short)0, acc0, false, false);
        acc1 = __builtin_amdgcn_wmma_f32_16x16x32_bf16(false, af, false, b1,
                 (short)0, acc1, false, false);
        acc2 = __builtin_amdgcn_wmma_f32_16x16x32_bf16(false, af, false, b2,
                 (short)0, acc2, false, false);
        acc3 = __builtin_amdgcn_wmma_f32_16x16x32_bf16(false, af, false, b3,
                 (short)0, acc3, false, false);
        acc4 = __builtin_amdgcn_wmma_f32_16x16x32_bf16(false, af, false, b4,
                 (short)0, acc4, false, false);
      }
      // elementwise cell, all in registers. D layout: VGPR r -> M = khalf*8+r,
      // N = lane&15 (CDNA5 ISA 32-bit C/D 16x16 layout).
      #pragma unroll
      for (int r = 0; r < 8; ++r) {
        int row = mt * 16 + khalf * 8 + r;
        float iv = acc0[r] + bi;
        float fl = acc1[r] + bfl;
        float fr = acc2[r] + bfr;
        float ov = acc3[r] + bo;
        float uv = acc4[r] + bu;
        int ls = lsel[row], rs = rsel[row];
        float cl = (ls >= 0) ? c_g[((size_t)ls * NB + row) * HD + colh] : 0.0f;
        float cr = (rs >= 0) ? c_g[((size_t)rs * NB + row) * HD + colh] : 0.0f;
        float c = sigf(iv) * tanhf(uv) + sigf(fl) * cl + sigf(fr) * cr;
        float h = sigf(ov) * tanhf(c);
        c_g[((size_t)t * NB + row) * HD + colh] = c;
        h_g[((size_t)t * NB + row) * HD + colh] = (__bf16)h;
        if (t == D_STEPS - 1) {
          outc[row * HD + colh] = c;
          outh[row * HD + colh] = h;
        }
      }
    }
    __threadfence();   // make h/c/idx stores visible before next step's gathers
    __syncthreads();
  }
}

extern "C" void kernel_launch(void* const* d_in, const int* in_sizes, int n_in,
                              void* d_out, int out_size, void* d_ws, size_t ws_size,
                              hipStream_t stream) {
  const int*   trans  = (const int*)d_in[0];    // [512,256] i32
  const int*   labels = (const int*)d_in[1];    // [512,256] i32
  const float* emb    = (const float*)d_in[2];  // [32000,64] f32
  const float* W      = (const float*)d_in[3];  // [320,640] f32
  const float* bias   = (const float*)d_in[4];  // [640] f32
  // d_in[5] = leaf placeholder (zeros) — folded into the sel<0 zero path.

  char* ws = (char*)d_ws;
  __bf16* Wp   = (__bf16*)(ws + WP_OFF);
  __bf16* embb = (__bf16*)(ws + EMB_OFF);
  __bf16* h_g  = (__bf16*)(ws + HG_OFF);
  float*  c_g  = (float*)(ws + CG_OFF);
  int*    idx  = (int*)(ws + IDX_OFF);

  // one-time setup (deterministic, full-chip parallel)
  spinn_pack_w<<<(NKT * NNT * 32 * 16 + 255) / 256, 256, 0, stream>>>(W, Wp);
  spinn_cvt_emb<<<(VOCAB * LD / 4 + 255) / 256, 256, 0, stream>>>(emb, embb);

  // persistent single-workgroup scan: 16 waves, ~167KB LDS for A + selectors
  size_t lds = (size_t)NB * ASTRIDE * 2 + 3 * NB * sizeof(int);
  spinn_persistent<<<1, 512, lds, stream>>>(trans, labels, embb, bias, Wp,
                                            h_g, c_g, idx, (float*)d_out);
}
